// BlockCausalSA_78340203479476
// MI455X (gfx1250) — matrix operand order repaired and checked by reading, not compiled, not run on hardware
//
#include <hip/hip_runtime.h>
#include <hip/hip_bf16.h>

// ---------- types ----------
typedef __bf16 bf16_t;
typedef __attribute__((ext_vector_type(16))) __bf16 v16bf;
typedef __attribute__((ext_vector_type(8)))  __bf16 v8bf;
typedef __attribute__((ext_vector_type(8)))  float  v8f;
typedef __attribute__((ext_vector_type(4)))  unsigned int u32x4;
typedef __attribute__((ext_vector_type(8)))  int i32x8;
typedef __attribute__((ext_vector_type(4)))  int i32x4;

#define B_   4
#define T_   4096
#define C_   1024
#define H_   16
#define HD_  64
#define LS_  256
#define NG_  (T_ / LS_)      // 16
#define M1_  (B_ * T_)       // 16384
#define N3C_ (3 * C_)        // 3072

// ---------- CDNA5 TDM (Tensor Data Mover) support ----------
#if defined(__has_builtin)
#  if __has_builtin(__builtin_amdgcn_tensor_load_to_lds)
#    define USE_TDM 1
#  endif
#endif
#ifndef USE_TDM
#  define USE_TDM 0
#endif

__device__ __forceinline__ void wait_tensorcnt0() {
#if defined(__has_builtin) && __has_builtin(__builtin_amdgcn_s_wait_tensorcnt)
  __builtin_amdgcn_s_wait_tensorcnt(0);
#else
  asm volatile("s_wait_tensorcnt 0x0" ::: "memory");
#endif
}

#if USE_TDM
// DMA a 2D tile of bf16 (tile_w elems x tile_h rows, row stride `stride` elems)
// from global memory into LDS at byte offset lds_addr, with optional LDS row
// padding: after (1<<(pad_interval_code+1)) DWORDs, insert (pad_amount_code+1)
// DWORDs of pad. D# layout per CDNA5 ISA ch.8 (group0 128b, group1 256b,
// groups2/3 zero = 2D tensor).  clang-23 toolchain: 6-arg builtin form.
__device__ __forceinline__ void tdm_load_2d(unsigned lds_addr, const void* gptr,
                                            unsigned tile_w, unsigned tile_h,
                                            unsigned stride,
                                            unsigned pad_interval_code,
                                            unsigned pad_amount_code,
                                            unsigned pad_enable) {
  unsigned long long ga = (unsigned long long)(uintptr_t)gptr;
  u32x4 g0;
  g0[0] = 1u;                                        // count=1, user mode
  g0[1] = lds_addr;                                  // lds_addr [63:32]
  g0[2] = (unsigned)(ga & 0xffffffffu);              // global_addr lo
  g0[3] = (unsigned)((ga >> 32) & 0x01ffffffu) | (2u << 30);  // addr hi | type=2
  i32x8 g1;
  g1[0] = (int)((1u << 16) |                         // data_size = 2 bytes
                (pad_enable << 20) |
                (pad_interval_code << 22) |
                (pad_amount_code << 25));
  g1[1] = (int)((tile_w & 0xffffu) << 16);           // tensor_dim0[15:0]
  g1[2] = (int)(((tile_w >> 16) & 0xffffu) |         // tensor_dim0[31:16]
                ((tile_h & 0xffffu) << 16));         // tensor_dim1[15:0]
  g1[3] = (int)(((tile_h >> 16) & 0xffffu) |         // tensor_dim1[31:16]
                ((tile_w & 0xffffu) << 16));         // tile_dim0
  g1[4] = (int)(tile_h & 0xffffu);                   // tile_dim1 (tile_dim2=0)
  g1[5] = (int)stride;                               // tensor_dim0_stride lo
  g1[6] = 0;                                         // stride hi / dim1_stride lo
  g1[7] = 0;
  i32x4 z4 = {0, 0, 0, 0};
  i32x8 z8 = {0, 0, 0, 0, 0, 0, 0, 0};
  __builtin_amdgcn_tensor_load_to_lds(g0, g1, z4, z4, z8, 0);
}
#endif

__device__ __forceinline__ v8f wmma_bf16(v16bf a, v16bf b, v8f c) {
  // D = A(16x32 bf16) * B(32x16 bf16) + C(16x16 f32)
  return __builtin_amdgcn_wmma_f32_16x16x32_bf16(false, a, false, b, (short)0, c,
                                                 false, false);
}

// build a 16-element bf16 fragment from two contiguous 8-element chunks
__device__ __forceinline__ v16bf ld_frag(const bf16_t* p0, const bf16_t* p1) {
  v8bf lo = *(const v8bf*)p0;
  v8bf hi = *(const v8bf*)p1;
  return __builtin_shufflevector(lo, hi, 0, 1, 2, 3, 4, 5, 6, 7,
                                 8, 9, 10, 11, 12, 13, 14, 15);
}

// ---------- fp32 -> bf16 conversion ----------
__global__ __launch_bounds__(256) void cvt_f32_bf16(const float* __restrict__ in,
                                                    bf16_t* __restrict__ out, int n) {
  int i = blockIdx.x * blockDim.x + threadIdx.x;
  int stride = gridDim.x * blockDim.x;
  for (; i < n; i += stride) out[i] = (bf16_t)in[i];
}

// ---------- bf16 GEMM: C[MxN] = A[MxK] * B[KxN], 128x128x32 tiles ----------
// Double-buffered: TDM DMA of the next A tile + vector staging of the next B
// tile overlap the WMMA work on the current tile. One barrier per iteration.
template <int OUT_F32>
__global__ __launch_bounds__(256) void gemm_bf16_k(const bf16_t* __restrict__ A,
                                                   const bf16_t* __restrict__ Bm,
                                                   void* __restrict__ Cp,
                                                   int M, int N, int K) {
  constexpr int LDA = 40;  // 16 DW of data + 4 DW pad per row
  constexpr int LDB = 40;
  __shared__ bf16_t sA[2][128 * LDA];   // [m][k], double buffered
  __shared__ bf16_t sB[2][128 * LDB];   // transposed [n][k], double buffered

  const int tid  = threadIdx.x;
  const int lane = tid & 31;
  const int wave = tid >> 5;
  const int hf   = lane >> 4;    // wave32 half
  const int l15  = lane & 15;
  const int wm   = wave >> 1;    // 0..3 (M)
  const int wn   = wave & 1;     // 0..1 (N)
  const int m0   = blockIdx.y * 128;
  const int n0   = blockIdx.x * 128;
  const int nk   = K / 32;

  v8f acc[2][4];
#pragma unroll
  for (int i = 0; i < 2; ++i)
#pragma unroll
    for (int j = 0; j < 4; ++j) acc[i][j] = v8f{};

  const int arow = tid >> 1, akc = (tid & 1) * 16;   // A-tile loader coords
  const int bkk = tid & 31, bnc = (tid >> 5) * 16;   // B-tile loader coords

  auto stageA = [&](int k0, int buf) {
#if USE_TDM
    if (tid < 32)  // one TDM issue from wave 0; pad 16DW+4DW -> LDA=40 bf16
      tdm_load_2d((unsigned)(uintptr_t)&sA[buf][0], A + (size_t)m0 * K + k0,
                  /*w*/32, /*h*/128, (unsigned)K, /*pi:16DW*/3, /*pa:4DW*/3, 1);
#else
    const bf16_t* g = A + (size_t)(m0 + arow) * K + k0 + akc;
    *(v8bf*)&sA[buf][arow * LDA + akc]     = *(const v8bf*)g;
    *(v8bf*)&sA[buf][arow * LDA + akc + 8] = *(const v8bf*)(g + 8);
#endif
  };
  auto stageB = [&](int k0, int buf) {
    const bf16_t* g = Bm + (size_t)(k0 + bkk) * N + n0 + bnc;
    v8bf b0 = *(const v8bf*)g;
    v8bf b1 = *(const v8bf*)(g + 8);
#pragma unroll
    for (int i = 0; i < 8; ++i) sB[buf][(bnc + i) * LDB + bkk] = b0[i];
#pragma unroll
    for (int i = 0; i < 8; ++i) sB[buf][(bnc + 8 + i) * LDB + bkk] = b1[i];
  };

  // prologue: stage tile 0
  stageA(0, 0);
  stageB(0, 0);

  for (int k = 0; k < nk; ++k) {
    const int cur = k & 1;
#if USE_TDM
    if (tid < 32) wait_tensorcnt0();  // TDM for tile k done (issuing wave)
#endif
    __syncthreads();                   // tile k visible to all waves
    if (k + 1 < nk) {                  // kick off tile k+1 into other buffer
      stageA((k + 1) * 32, cur ^ 1);
      stageB((k + 1) * 32, cur ^ 1);
    }

    v16bf af[2], bfg[4];
#pragma unroll
    for (int i = 0; i < 2; ++i) {
      const bf16_t* r = &sA[cur][(wm * 32 + i * 16 + l15) * LDA];
      af[i] = ld_frag(r + hf * 8, r + 16 + hf * 8);
    }
#pragma unroll
    for (int j = 0; j < 4; ++j) {
      const bf16_t* r = &sB[cur][(wn * 64 + j * 16 + l15) * LDB];
      bfg[j] = ld_frag(r + hf * 8, r + 16 + hf * 8);
    }
#pragma unroll
    for (int i = 0; i < 2; ++i)
#pragma unroll
      for (int j = 0; j < 4; ++j) acc[i][j] = wmma_bf16(af[i], bfg[j], acc[i][j]);
  }

  // epilogue: D layout = VGPR v -> row hf*8+v, col l15
#pragma unroll
  for (int i = 0; i < 2; ++i) {
    const int m = m0 + wm * 32 + i * 16 + hf * 8;
#pragma unroll
    for (int j = 0; j < 4; ++j) {
      const int n = n0 + wn * 64 + j * 16 + l15;
#pragma unroll
      for (int v = 0; v < 8; ++v) {
        float val = acc[i][j][v];
        if (OUT_F32)
          ((float*)Cp)[(size_t)(m + v) * N + n] = val;
        else
          ((bf16_t*)Cp)[(size_t)(m + v) * N + n] = (bf16_t)val;
      }
    }
  }
}

// ---------- RMSNorm on q & k rows of qkv (in place, bf16) ----------
__global__ __launch_bounds__(256) void rmsnorm_qk_k(bf16_t* __restrict__ qkv,
                                                    const float* __restrict__ w) {
  const int rowIdx = blockIdx.x >> 1;
  const int which  = blockIdx.x & 1;  // 0=q, 1=k
  bf16_t* base = qkv + (size_t)rowIdx * N3C_ + which * C_;
  const int tid = threadIdx.x;

  float vals[4];
  float ss = 0.f;
#pragma unroll
  for (int i = 0; i < 4; ++i) {
    float f = (float)base[tid + i * 256];
    vals[i] = f;
    ss += f * f;
  }
#pragma unroll
  for (int m = 1; m <= 16; m <<= 1) ss += __shfl_xor(ss, m, 32);
  __shared__ float sred[8];
  if ((tid & 31) == 0) sred[tid >> 5] = ss;
  __syncthreads();
  float tot = 0.f;
#pragma unroll
  for (int i = 0; i < 8; ++i) tot += sred[i];
  const float scale = rsqrtf(tot * (1.0f / C_) + 1e-6f);
#pragma unroll
  for (int i = 0; i < 4; ++i) {
    int c = tid + i * 256;
    base[c] = (bf16_t)(vals[i] * scale * w[c]);
  }
}

// ---------- block-causal flash attention: one WG per (b, group, head) ----------
__global__ __launch_bounds__(256) void attn_k(const bf16_t* __restrict__ qkv,
                                              bf16_t* __restrict__ y) {
  constexpr int LDK = 72;    // 32 DW data + 4 DW pad (matches TDM pad config)
  constexpr int LDV = 264;   // sVt row stride
  constexpr int LDP = 40;    // sP row stride
  __shared__ bf16_t sK[LS_ * LDK];     // K rows [key][hd]      (36,864 B)
  __shared__ bf16_t sVt[HD_ * LDV];    // V transposed [hd][key](33,792 B)
  __shared__ bf16_t sP[8 * 32 * LDP];  // per-wave P bounce     (20,480 B)

  const int wg  = blockIdx.x;
  const int b   = wg / (NG_ * H_);
  const int rem = wg % (NG_ * H_);
  const int g   = rem / H_;
  const int h   = rem % H_;
  const size_t rowbase = (size_t)(b * T_ + g * LS_);
  const bf16_t* Qg = qkv + rowbase * N3C_ + h * HD_;
  const bf16_t* Kg = Qg + C_;
  const bf16_t* Vg = Qg + 2 * C_;

  const int tid  = threadIdx.x;
  const int lane = tid & 31;
  const int wave = tid >> 5;   // 0..7, owns query rows [wave*32, wave*32+32)
  const int hf   = lane >> 4;
  const int l15  = lane & 15;

  // kick off K-tile DMA first (256 rows x 64 bf16, stride 3072, pad 32DW+4DW),
  // then overlap it with V-transpose staging and Q fragment loads.
#if USE_TDM
  if (tid < 32)
    tdm_load_2d((unsigned)(uintptr_t)&sK[0], Kg,
                /*w*/HD_, /*h*/LS_, /*stride*/N3C_, /*pi:32DW*/4, /*pa:4DW*/3, 1);
#else
  {
    const bf16_t* kr = Kg + (size_t)tid * N3C_;
    bf16_t* kd = &sK[tid * LDK];
#pragma unroll
    for (int i = 0; i < 8; ++i) *(v8bf*)(kd + i * 8) = *(const v8bf*)(kr + i * 8);
  }
#endif
  // stage V transposed (manual: TDM has no transpose)
  {
    const bf16_t* vr = Vg + (size_t)tid * N3C_;
#pragma unroll
    for (int i = 0; i < 8; ++i) {
      v8bf vv = *(const v8bf*)(vr + i * 8);
#pragma unroll
      for (int e = 0; e < 8; ++e) sVt[(i * 8 + e) * LDV + tid] = vv[e];
    }
  }

  // Q fragments for this wave's 32 rows (2 row-tiles x 2 hd-chunks of 32)
  v16bf qf[2][2];
#pragma unroll
  for (int i = 0; i < 2; ++i) {
    const bf16_t* qr = Qg + (size_t)(wave * 32 + i * 16 + l15) * N3C_;
#pragma unroll
    for (int hc = 0; hc < 2; ++hc)
      qf[i][hc] = ld_frag(qr + hc * 32 + hf * 8, qr + hc * 32 + 16 + hf * 8);
  }
#if USE_TDM
  if (tid < 32) wait_tensorcnt0();  // issuing wave drains TDM before barrier
#endif
  __syncthreads();

  v8f acc[2][4];
#pragma unroll
  for (int r = 0; r < 2; ++r)
#pragma unroll
    for (int n = 0; n < 4; ++n) acc[r][n] = v8f{};
  float mrow[2][8], lrow[2][8];
#pragma unroll
  for (int r = 0; r < 2; ++r)
#pragma unroll
    for (int v = 0; v < 8; ++v) { mrow[r][v] = -3.0e38f; lrow[r][v] = 0.f; }

  bf16_t* pw = &sP[wave * 32 * LDP];

  // triangular key loop: chunk kt covers keys [kt*32, kt*32+32)
  for (int kt = 0; kt <= wave; ++kt) {
    // S = Q K^T; 2x2 tiles of 16x16
    v8f s[2][2];
#pragma unroll
    for (int c = 0; c < 2; ++c) {
      const bf16_t* krow = &sK[(kt * 32 + c * 16 + l15) * LDK];
      v16bf kf0 = ld_frag(krow + hf * 8, krow + 16 + hf * 8);
      v16bf kf1 = ld_frag(krow + 32 + hf * 8, krow + 32 + 16 + hf * 8);
#pragma unroll
      for (int r = 0; r < 2; ++r) {
        v8f z{};
        z = wmma_bf16(qf[r][0], kf0, z);
        s[r][c] = wmma_bf16(qf[r][1], kf1, z);
      }
    }
    // scale + causal mask (mask only on the diagonal chunk kt == wave)
#pragma unroll
    for (int r = 0; r < 2; ++r)
#pragma unroll
      for (int c = 0; c < 2; ++c)
#pragma unroll
        for (int v = 0; v < 8; ++v) {
          float val = s[r][c][v] * 0.125f;  // 1/sqrt(64)
          if (kt == wave && (c * 16 + l15) > (r * 16 + hf * 8 + v)) val = -3.0e38f;
          s[r][c][v] = val;
        }
    // online softmax update per row (rows live in 16-lane halves)
#pragma unroll
    for (int r = 0; r < 2; ++r) {
      float alpha[8];
#pragma unroll
      for (int v = 0; v < 8; ++v) {
        float mx = fmaxf(s[r][0][v], s[r][1][v]);
#pragma unroll
        for (int m = 1; m <= 8; m <<= 1) mx = fmaxf(mx, __shfl_xor(mx, m, 32));
        float mnew = fmaxf(mrow[r][v], mx);
        float a  = __expf(mrow[r][v] - mnew);
        float p0 = __expf(s[r][0][v] - mnew);
        float p1 = __expf(s[r][1][v] - mnew);
        s[r][0][v] = p0; s[r][1][v] = p1;
        float ps = p0 + p1;
#pragma unroll
        for (int m = 1; m <= 8; m <<= 1) ps += __shfl_xor(ps, m, 32);
        lrow[r][v] = lrow[r][v] * a + ps;
        mrow[r][v] = mnew;
        alpha[v] = a;
      }
#pragma unroll
      for (int n = 0; n < 4; ++n)
#pragma unroll
        for (int v = 0; v < 8; ++v) acc[r][n][v] *= alpha[v];
      // bounce P through LDS: C-layout -> A-layout (intra-wave only)
#pragma unroll
      for (int c = 0; c < 2; ++c)
#pragma unroll
        for (int v = 0; v < 8; ++v)
          pw[(r * 16 + hf * 8 + v) * LDP + c * 16 + l15] = (bf16_t)s[r][c][v];
    }
    asm volatile("s_wait_dscnt 0x0" ::: "memory");  // DS in-order within wave
    // O += P (16x32) @ V-chunk (32x64)
#pragma unroll
    for (int r = 0; r < 2; ++r) {
      const bf16_t* pr = &pw[(r * 16 + l15) * LDP];
      v16bf pf = ld_frag(pr + hf * 8, pr + 16 + hf * 8);
#pragma unroll
      for (int n = 0; n < 4; ++n) {
        const bf16_t* vrow = &sVt[(n * 16 + l15) * LDV + kt * 32];
        v16bf vf = ld_frag(vrow + hf * 8, vrow + 16 + hf * 8);
        acc[r][n] = wmma_bf16(pf, vf, acc[r][n]);
      }
    }
  }

  // epilogue: y[b, t, h*64 + d] = acc / l
#pragma unroll
  for (int r = 0; r < 2; ++r)
#pragma unroll
    for (int n = 0; n < 4; ++n)
#pragma unroll
      for (int v = 0; v < 8; ++v) {
        int qrow = wave * 32 + r * 16 + hf * 8 + v;
        size_t t = rowbase + qrow;
        int col = h * HD_ + n * 16 + l15;
        y[t * C_ + col] = (bf16_t)(acc[r][n][v] / lrow[r][v]);
      }
}

// ---------- launcher ----------
extern "C" void kernel_launch(void* const* d_in, const int* in_sizes, int n_in,
                              void* d_out, int out_size, void* d_ws, size_t ws_size,
                              hipStream_t stream) {
  const float* x      = (const float*)d_in[0];
  const float* w_qkv  = (const float*)d_in[1];
  const float* ln_w   = (const float*)d_in[2];
  const float* w_proj = (const float*)d_in[3];
  float* out = (float*)d_out;

  // workspace carve-up (~176 MB total)
  char* ws = (char*)d_ws;
  bf16_t* x_bf    = (bf16_t*)ws; ws += (size_t)M1_ * C_ * sizeof(bf16_t);
  bf16_t* wqkv_bf = (bf16_t*)ws; ws += (size_t)C_ * N3C_ * sizeof(bf16_t);
  bf16_t* wproj_bf= (bf16_t*)ws; ws += (size_t)C_ * C_ * sizeof(bf16_t);
  bf16_t* qkv     = (bf16_t*)ws; ws += (size_t)M1_ * N3C_ * sizeof(bf16_t);
  bf16_t* ybf     = (bf16_t*)ws; ws += (size_t)M1_ * C_ * sizeof(bf16_t);
  (void)ws_size; (void)n_in; (void)in_sizes; (void)out_size;

  auto cvtLaunch = [&](const float* src, bf16_t* dst, int n) {
    int blocks = (n / 4 + 255) / 256;
    cvt_f32_bf16<<<blocks, 256, 0, stream>>>(src, dst, n);
  };
  cvtLaunch(x, x_bf, M1_ * C_);
  cvtLaunch(w_qkv, wqkv_bf, C_ * N3C_);
  cvtLaunch(w_proj, wproj_bf, C_ * C_);

  // qkv = x @ w_qkv  (bf16 out)
  gemm_bf16_k<0><<<dim3(N3C_ / 128, M1_ / 128), 256, 0, stream>>>(
      x_bf, wqkv_bf, (void*)qkv, M1_, N3C_, C_);

  // RMSNorm q,k in place
  rmsnorm_qk_k<<<M1_ * 2, 256, 0, stream>>>(qkv, ln_w);

  // block-causal attention -> ybf
  attn_k<<<B_ * NG_ * H_, 256, 0, stream>>>(qkv, ybf);

  // out = y @ w_proj (f32 out)
  gemm_bf16_k<1><<<dim3(C_ / 128, M1_ / 128), 256, 0, stream>>>(
      ybf, wproj_bf, (void*)out, M1_, C_, C_);
}
